// MultiHeadAttention_59863254171746
// MI455X (gfx1250) — compile-verified, matrix operand order checked by reading, not run on hardware
//
#include <hip/hip_runtime.h>
#include <hip/hip_bf16.h>
#include <stdint.h>

// ---------------------------------------------------------------------------
// MI455X (gfx1250) multi-head attention forward, bf16 WMMA pipeline.
// - async global->LDS staging (ASYNCcnt) for pure-bf16 tiles
// - double-buffered LDS tiles, one barrier per k-iteration
// - LDS leading dim padded 64->72 to kill ds_load_b128 bank conflicts
// B=4, S=2048, D=1024, H=16, DK=DV=64.
// ---------------------------------------------------------------------------

typedef __bf16 bf16;
typedef __bf16 v16bf  __attribute__((ext_vector_type(16)));
typedef __bf16 bf16x8 __attribute__((ext_vector_type(8)));
typedef float  v8f    __attribute__((ext_vector_type(8)));
typedef float  f32x4  __attribute__((ext_vector_type(4)));

static constexpr int MDIM = 8192;   // B*S
static constexpr int KDIM = 1024;   // D
static constexpr int NDIM = 1024;   // H*DK
static constexpr int SEQ  = 2048;
static constexpr int NH   = 16;
static constexpr int LDA  = 72;                 // padded LDS stride (bf16 elems)
static constexpr int TILE_B = 128 * LDA * 2;    // bytes per 128x64 padded tile

extern __shared__ char dynsmem[];

// LDS buffer selectors (no pointer arrays -> no static addrspacecast inits)
#define GEMM_SA(i) ((bf16*)(dynsmem + (i) * TILE_B))
#define GEMM_SB(i) ((bf16*)(dynsmem + (2 + (i)) * TILE_B))

__device__ __forceinline__ v8f zero8() {
  v8f z = {0.f, 0.f, 0.f, 0.f, 0.f, 0.f, 0.f, 0.f};
  return z;
}

__device__ __forceinline__ v8f wmma_bf16(v16bf a, v16bf b, v8f c) {
  return __builtin_amdgcn_wmma_f32_16x16x32_bf16(
      /*neg_a=*/false, a, /*neg_b=*/false, b,
      /*c_mod=*/(short)0, c, /*reuse_a=*/false, /*reuse_b=*/false);
}

// Async 16-byte global->LDS copy (ASYNCcnt-tracked, no VGPR round trip).
__device__ __forceinline__ void async_copy16(void* lds_dst, const void* gsrc) {
  uint32_t l = (uint32_t)(uintptr_t)lds_dst;
  uint64_t g = (uint64_t)(uintptr_t)gsrc;
  asm volatile("global_load_async_to_lds_b128 %0, %1, off"
               :: "v"(l), "v"(g)
               : "memory");
}
__device__ __forceinline__ void wait_async() {
  asm volatile("s_wait_asynccnt 0x0" ::: "memory");
}

// 16x32 fragment load from LDS (A-layout, or B from an N-major tile).
// lanes 0-15: row lane,    K {0..7,16..23}
// lanes 16-31: row lane-16, K {8..15,24..31}
__device__ __forceinline__ v16bf load_frag(const bf16* base, int ld, int lane) {
  const int r  = lane & 15;
  const int kb = (lane >> 4) * 8;
  const bf16* p = base + r * ld + kb;
  bf16x8 lo = *(const bf16x8*)(p);
  bf16x8 hi = *(const bf16x8*)(p + 16);
  v16bf f;
#pragma unroll
  for (int i = 0; i < 8; ++i) { f[i] = lo[i]; f[i + 8] = hi[i]; }
  return f;
}

// ---------------------------------------------------------------------------
// GEMM 1: Y_headmajor(bf16) = (A(f32)[8192x1024] @ W(f32)[1024x1024] + bias)*scale
// Double-buffered, BK=64, one barrier per k-iteration.
// ---------------------------------------------------------------------------
__global__ void __launch_bounds__(256)
gemm_proj_kernel(const float* __restrict__ A, const float* __restrict__ W,
                 const float* __restrict__ bias, bf16* __restrict__ Y,
                 float scale) {
  const int t    = threadIdx.x;
  const int lane = t & 31;
  const int wave = t >> 5;
  const int wr   = wave >> 1;
  const int wc   = wave & 1;
  const int m0   = blockIdx.x * 128;
  const int n0   = blockIdx.y * 128;

  v8f acc[2][4];
#pragma unroll
  for (int i = 0; i < 2; ++i)
#pragma unroll
    for (int j = 0; j < 4; ++j) acc[i][j] = zero8();

  const int arow = t >> 1;
  const int acol = (t & 1) * 32;
  const int bn   = t & 127;
  const int bk0  = (t >> 7) * 32;

  auto stageA = [&](int kt, bf16* dst) {
    const float* ga = A + (size_t)(m0 + arow) * KDIM + kt * 64 + acol;
#pragma unroll
    for (int u = 0; u < 4; ++u) {
      f32x4 f0 = *(const f32x4*)(ga + u * 8);
      f32x4 f1 = *(const f32x4*)(ga + u * 8 + 4);
      bf16x8 hv;
#pragma unroll
      for (int i = 0; i < 4; ++i) { hv[i] = (bf16)f0[i]; hv[i + 4] = (bf16)f1[i]; }
      *(bf16x8*)&dst[arow * LDA + acol + u * 8] = hv;
    }
  };
  auto stageB = [&](int kt, bf16* dst) {
#pragma unroll
    for (int u = 0; u < 4; ++u) {
      bf16x8 hv;
#pragma unroll
      for (int i = 0; i < 8; ++i) {
        float wv = W[(size_t)(kt * 64 + bk0 + u * 8 + i) * NDIM + n0 + bn];
        hv[i] = (bf16)wv;
      }
      *(bf16x8*)&dst[bn * LDA + bk0 + u * 8] = hv;
    }
  };

  stageA(0, GEMM_SA(0));
  stageB(0, GEMM_SB(0));

  for (int kt = 0; kt < KDIM / 64; ++kt) {
    __syncthreads();  // buf[kt&1] staged; other buf free to overwrite
    const int cur = kt & 1;
    if (kt + 1 < KDIM / 64) {
      stageA(kt + 1, GEMM_SA(1 - cur));
      stageB(kt + 1, GEMM_SB(1 - cur));
      __builtin_prefetch(&W[(size_t)((kt + 1) * 64 + bk0) * NDIM + n0 + bn], 0, 1);
    }
    const bf16* sAc = GEMM_SA(cur);
    const bf16* sBc = GEMM_SB(cur);
#pragma unroll
    for (int d0 = 0; d0 < 64; d0 += 32) {
      v16bf af[2], bfr[4];
#pragma unroll
      for (int i = 0; i < 2; ++i)
        af[i] = load_frag(&sAc[(wr * 32 + i * 16) * LDA + d0], LDA, lane);
#pragma unroll
      for (int j = 0; j < 4; ++j)
        bfr[j] = load_frag(&sBc[(wc * 64 + j * 16) * LDA + d0], LDA, lane);
#pragma unroll
      for (int i = 0; i < 2; ++i)
#pragma unroll
        for (int j = 0; j < 4; ++j)
          acc[i][j] = wmma_bf16(af[i], bfr[j], acc[i][j]);
    }
  }

  const int cl     = lane & 15;
  const int rowoff = (lane >> 4) * 8;
#pragma unroll
  for (int i = 0; i < 2; ++i)
#pragma unroll
    for (int j = 0; j < 4; ++j) {
      const int n  = n0 + wc * 64 + j * 16 + cl;
      const float bb = bias[n];
      const int h  = n >> 6;
      const int dk = n & 63;
#pragma unroll
      for (int e = 0; e < 8; ++e) {
        const int m = m0 + wr * 32 + i * 16 + rowoff + e;
        const int b = m >> 11;
        const int s = m & 2047;
        const size_t dst = ((size_t)((b * NH + h) * SEQ + s)) * 64 + dk;
        Y[dst] = (bf16)((acc[i][j][e] + bb) * scale);
      }
    }
}

// ---------------------------------------------------------------------------
// GEMM 2: out(f32) = O(bf16)[8192x1024] @ Wo(f32)[1024x1024] + bo
// A tile double-buffered via async global->LDS copies.
// ---------------------------------------------------------------------------
__global__ void __launch_bounds__(256)
gemm_out_kernel(const bf16* __restrict__ A, const float* __restrict__ W,
                const float* __restrict__ bias, float* __restrict__ out) {
  const int t    = threadIdx.x;
  const int lane = t & 31;
  const int wave = t >> 5;
  const int wr   = wave >> 1;
  const int wc   = wave & 1;
  const int m0   = blockIdx.x * 128;
  const int n0   = blockIdx.y * 128;

  v8f acc[2][4];
#pragma unroll
  for (int i = 0; i < 2; ++i)
#pragma unroll
    for (int j = 0; j < 4; ++j) acc[i][j] = zero8();

  const int arow = t >> 1;
  const int acol = (t & 1) * 32;
  const int bn   = t & 127;
  const int bk0  = (t >> 7) * 32;

  auto stageA = [&](int kt, bf16* dst) {   // async, pure bf16 copy
    const bf16* ga = A + (size_t)(m0 + arow) * KDIM + kt * 64 + acol;
    bf16* la = &dst[arow * LDA + acol];
#pragma unroll
    for (int u = 0; u < 4; ++u)
      async_copy16(la + u * 8, ga + u * 8);
  };
  auto stageB = [&](int kt, bf16* dst) {
#pragma unroll
    for (int u = 0; u < 4; ++u) {
      bf16x8 hv;
#pragma unroll
      for (int i = 0; i < 8; ++i) {
        float wv = W[(size_t)(kt * 64 + bk0 + u * 8 + i) * NDIM + n0 + bn];
        hv[i] = (bf16)wv;
      }
      *(bf16x8*)&dst[bn * LDA + bk0 + u * 8] = hv;
    }
  };

  stageA(0, GEMM_SA(0));
  stageB(0, GEMM_SB(0));

  for (int kt = 0; kt < KDIM / 64; ++kt) {
    wait_async();     // copies into buf[kt&1] complete (this wave)
    __syncthreads();  // all waves staged; prior readers of other buf done
    const int cur = kt & 1;
    if (kt + 1 < KDIM / 64) {
      stageA(kt + 1, GEMM_SA(1 - cur));
      stageB(kt + 1, GEMM_SB(1 - cur));
      __builtin_prefetch(&W[(size_t)((kt + 1) * 64 + bk0) * NDIM + n0 + bn], 0, 1);
    }
    const bf16* sAc = GEMM_SA(cur);
    const bf16* sBc = GEMM_SB(cur);
#pragma unroll
    for (int d0 = 0; d0 < 64; d0 += 32) {
      v16bf af[2], bfr[4];
#pragma unroll
      for (int i = 0; i < 2; ++i)
        af[i] = load_frag(&sAc[(wr * 32 + i * 16) * LDA + d0], LDA, lane);
#pragma unroll
      for (int j = 0; j < 4; ++j)
        bfr[j] = load_frag(&sBc[(wc * 64 + j * 16) * LDA + d0], LDA, lane);
#pragma unroll
      for (int i = 0; i < 2; ++i)
#pragma unroll
        for (int j = 0; j < 4; ++j)
          acc[i][j] = wmma_bf16(af[i], bfr[j], acc[i][j]);
    }
  }

  const int cl     = lane & 15;
  const int rowoff = (lane >> 4) * 8;
#pragma unroll
  for (int i = 0; i < 2; ++i)
#pragma unroll
    for (int j = 0; j < 4; ++j) {
      const int n  = n0 + wc * 64 + j * 16 + cl;
      const float bb = bias[n];
#pragma unroll
      for (int e = 0; e < 8; ++e) {
        const int m = m0 + wr * 32 + i * 16 + rowoff + e;
        out[(size_t)m * NDIM + n] = acc[i][j][e] + bb;
      }
    }
}

// ---------------------------------------------------------------------------
// Attention: one workgroup per (b, h, 128-row q block).
// Pass A: scores = qh @ kh^T; async double-buffered K stream; exp + row sums.
// Pass B: read exp-scores back, normalize in place (final attn), P@V via WMMA.
// ---------------------------------------------------------------------------
__global__ void __launch_bounds__(256)
attn_kernel(const bf16* __restrict__ qh, const bf16* __restrict__ kh,
            const bf16* __restrict__ vh, float* __restrict__ attn,
            bf16* __restrict__ obuf) {
  __shared__ __align__(16) char smemraw[3 * TILE_B + 512];
#define ATTN_SQ   ((bf16*)smemraw)
#define ATTN_SK(i) ((bf16*)(smemraw + (1 + (i)) * TILE_B))
#define ATTN_RS   ((float*)(smemraw + 3 * TILE_B))
  bf16*  sP = ATTN_SQ;       // pass B: 128xLDA
  bf16*  sV = ATTN_SK(0);    // pass B: 64xLDA (dv-major)

  const int t    = threadIdx.x;
  const int lane = t & 31;
  const int wave = t >> 5;
  const int wr   = wave >> 1;
  const int wc   = wave & 1;
  const int qb   = blockIdx.x;
  const int h    = blockIdx.y;
  const int b    = blockIdx.z;

  const size_t bh = (size_t)(b * NH + h);
  const bf16* qbase = qh + (bh * SEQ + (size_t)qb * 128) * 64;
  const bf16* kbase = kh + bh * SEQ * 64;
  const bf16* vbase = vh + bh * SEQ * 64;
  float* attnbase = attn + bh * SEQ * SEQ + (size_t)qb * 128 * SEQ;

  if (t < 128) ATTN_RS[t] = 0.0f;

  const int srow = t >> 1;
  const int soff = (t & 1) * 32;

  auto stageK = [&](int kb, bf16* dst) {   // async bf16 copy, 64B per thread
    const bf16* g = kbase + (size_t)(kb * 128 + srow) * 64 + soff;
    bf16* l = &dst[srow * LDA + soff];
#pragma unroll
    for (int i = 0; i < 4; ++i)
      async_copy16(l + i * 8, g + i * 8);
  };

  // async-stage Q (resident through pass A) + first K tile
  {
    const bf16* g = qbase + srow * 64 + soff;
    bf16* l = &ATTN_SQ[srow * LDA + soff];
#pragma unroll
    for (int i = 0; i < 4; ++i)
      async_copy16(l + i * 8, g + i * 8);
  }
  stageK(0, ATTN_SK(0));

  const int cl     = lane & 15;
  const int rowoff = (lane >> 4) * 8;

  float ps[16];
#pragma unroll
  for (int i = 0; i < 16; ++i) ps[i] = 0.f;

  // ---------------- Pass A ----------------
  for (int kb = 0; kb < SEQ / 128; ++kb) {
    wait_async();     // copies into sK[kb&1] (and sQ on first iter) done
    __syncthreads();
    const int cur = kb & 1;
    if (kb + 1 < SEQ / 128) stageK(kb + 1, ATTN_SK(1 - cur));

    v8f acc[2][4];
#pragma unroll
    for (int i = 0; i < 2; ++i)
#pragma unroll
      for (int j = 0; j < 4; ++j) acc[i][j] = zero8();

    const bf16* sKc = ATTN_SK(cur);
#pragma unroll
    for (int d0 = 0; d0 < 64; d0 += 32) {
      v16bf af[2], bfr[4];
#pragma unroll
      for (int i = 0; i < 2; ++i)
        af[i] = load_frag(&ATTN_SQ[(wr * 32 + i * 16) * LDA + d0], LDA, lane);
#pragma unroll
      for (int j = 0; j < 4; ++j)
        bfr[j] = load_frag(&sKc[(wc * 64 + j * 16) * LDA + d0], LDA, lane);
#pragma unroll
      for (int i = 0; i < 2; ++i)
#pragma unroll
        for (int j = 0; j < 4; ++j)
          acc[i][j] = wmma_bf16(af[i], bfr[j], acc[i][j]);
    }

#pragma unroll
    for (int i = 0; i < 2; ++i)
#pragma unroll
      for (int j = 0; j < 4; ++j) {
        const int kcol = kb * 128 + wc * 64 + j * 16 + cl;
#pragma unroll
        for (int e = 0; e < 8; ++e) {
          const int qr = wr * 32 + i * 16 + rowoff + e;
          const float ev = __expf(acc[i][j][e]);
          ps[i * 8 + e] += ev;
          attnbase[(size_t)qr * SEQ + kcol] = ev;
        }
      }
  }

#pragma unroll
  for (int i = 0; i < 2; ++i)
#pragma unroll
    for (int e = 0; e < 8; ++e)
      atomicAdd(&ATTN_RS[wr * 32 + i * 16 + rowoff + e], ps[i * 8 + e]);
  __threadfence();   // pass-A global stores visible to readback
  __syncthreads();   // rowsum complete

  // ---------------- Pass B ----------------
  v8f accD[4];
#pragma unroll
  for (int j = 0; j < 4; ++j) accD[j] = zero8();
  const int qw0 = wave * 16;

  for (int kc = 0; kc < SEQ / 64; ++kc) {
    __syncthreads();  // protect sP/sV reuse
    // stage V half-block transposed to [dv][kk], stride LDA
    {
      const int kk = t >> 2, dvb = (t & 3) * 16;
      const bf16* g = vbase + (size_t)(kc * 64 + kk) * 64 + dvb;
      bf16x8 a0 = *(const bf16x8*)(g);
      bf16x8 a1 = *(const bf16x8*)(g + 8);
#pragma unroll
      for (int i = 0; i < 8; ++i) {
        sV[(dvb + i) * LDA + kk]     = a0[i];
        sV[(dvb + 8 + i) * LDA + kk] = a1[i];
      }
    }
    // read exp-scores back, normalize in place, stage bf16 P tile
    {
      const float rinv = 1.0f / ATTN_RS[srow];
      float* gp = attnbase + (size_t)srow * SEQ + kc * 64 + soff;
#pragma unroll
      for (int u = 0; u < 4; ++u) {
        f32x4 f0 = *(const f32x4*)(gp + u * 8);
        f32x4 f1 = *(const f32x4*)(gp + u * 8 + 4);
        bf16x8 hp;
#pragma unroll
        for (int i = 0; i < 4; ++i) {
          f0[i] *= rinv; f1[i] *= rinv;
          hp[i] = (bf16)f0[i]; hp[4 + i] = (bf16)f1[i];
        }
        *(f32x4*)(gp + u * 8)     = f0;
        *(f32x4*)(gp + u * 8 + 4) = f1;
        *(bf16x8*)&sP[srow * LDA + soff + u * 8] = hp;
      }
    }
    __syncthreads();

#pragma unroll
    for (int ks = 0; ks < 64; ks += 32) {
      v16bf af = load_frag(&sP[qw0 * LDA + ks], LDA, lane);
#pragma unroll
      for (int j = 0; j < 4; ++j) {
        v16bf bfr = load_frag(&sV[(j * 16) * LDA + ks], LDA, lane);
        accD[j] = wmma_bf16(af, bfr, accD[j]);
      }
    }
  }

  // write O as bf16 in [b, s, h*64 + dv] layout
#pragma unroll
  for (int j = 0; j < 4; ++j) {
    const int dv = j * 16 + cl;
#pragma unroll
    for (int e = 0; e < 8; ++e) {
      const int qr = qb * 128 + qw0 + rowoff + e;
      const size_t dst = ((size_t)(b * SEQ + qr)) * 1024 + h * 64 + dv;
      obuf[dst] = (bf16)accD[j][e];
    }
  }
#undef ATTN_SQ
#undef ATTN_SK
#undef ATTN_RS
}

// ---------------------------------------------------------------------------
// Host-side launcher
// ---------------------------------------------------------------------------
extern "C" void kernel_launch(void* const* d_in, const int* in_sizes, int n_in,
                              void* d_out, int out_size, void* d_ws, size_t ws_size,
                              hipStream_t stream) {
  (void)in_sizes; (void)n_in; (void)out_size; (void)ws_size;

  const float* q  = (const float*)d_in[0];
  const float* k  = (const float*)d_in[1];
  const float* v  = (const float*)d_in[2];
  const float* Wq = (const float*)d_in[3];
  const float* bq = (const float*)d_in[4];
  const float* Wk = (const float*)d_in[5];
  const float* bk = (const float*)d_in[6];
  const float* Wv = (const float*)d_in[7];
  const float* bv = (const float*)d_in[8];
  const float* Wo = (const float*)d_in[9];
  const float* bo = (const float*)d_in[10];

  float* out  = (float*)d_out;                    // [8192, 1024]
  float* attn = out + (size_t)MDIM * NDIM;        // [4, 16, 2048, 2048]

  bf16* qh = (bf16*)d_ws;                         // [b,h,s,64] bf16
  bf16* kh = qh + (size_t)MDIM * NDIM;
  bf16* vh = kh + (size_t)MDIM * NDIM;
  bf16* ob = vh + (size_t)MDIM * NDIM;            // [b,s,h*64+dv] bf16

  const int gemm_lds = 4 * TILE_B;                // 73728 B (< 320 KB/WGP)
  (void)hipFuncSetAttribute((const void*)gemm_proj_kernel,
                            hipFuncAttributeMaxDynamicSharedMemorySize, gemm_lds);
  (void)hipFuncSetAttribute((const void*)gemm_out_kernel,
                            hipFuncAttributeMaxDynamicSharedMemorySize, gemm_lds);

  dim3 blk(256);
  dim3 gg(MDIM / 128, NDIM / 128);

  // q scaled by 1/sqrt(DK)=0.125 so scores come out pre-scaled
  gemm_proj_kernel<<<gg, blk, gemm_lds, stream>>>(q, Wq, bq, qh, 0.125f);
  gemm_proj_kernel<<<gg, blk, gemm_lds, stream>>>(k, Wk, bk, kh, 1.0f);
  gemm_proj_kernel<<<gg, blk, gemm_lds, stream>>>(v, Wv, bv, vh, 1.0f);

  attn_kernel<<<dim3(SEQ / 128, NH, 4), blk, 0, stream>>>(qh, kh, vh, attn, ob);

  gemm_out_kernel<<<gg, blk, 0, stream>>>(ob, Wo, bo, out);
}
